// SlotAttentionVideo_84791244357695
// MI455X (gfx1250) — compile-verified, matrix operand order checked by reading, not run on hardware
//
#include <hip/hip_runtime.h>
#include <stdint.h>

// ---------------------------------------------------------------------------
// CDNA5 (gfx1250) Slot-Attention-Video forward.
// Heavy math: implicit-GEMM with V_WMMA_F32_16X16X32_BF16 (wave32, 8 waves,
// 128x64 block tile, 32x32 per-wave tile, KB=64 -> 8 WMMAs per LDS stage,
// LDS double buffer, magic-number division for gather address math,
// weights pre-converted once to bf16 [n][Kpad], vectorized f32->bf16 A
// staging with uniform full-tile fast paths). Decoder evaluated only for the
// final timestep (reference discards earlier timesteps' decode).
// ---------------------------------------------------------------------------

typedef __bf16 bf16_t;
typedef bf16_t bf16x16 __attribute__((ext_vector_type(16)));
typedef float  f32x8   __attribute__((ext_vector_type(8)));

#define DEV __device__ __forceinline__

DEV uint16_t f2bf(float f) {
  union { float f; uint32_t u; } c; c.f = f;
  uint32_t u = c.u;
  uint32_t r = u + 0x7FFFu + ((u >> 16) & 1u);   // round-to-nearest-even
  return (uint16_t)(r >> 16);
}

DEV uint32_t pack2bf(float x, float y) {         // native cvt (RNE) + pack
  union { bf16_t h[2]; uint32_t u; } c;
  c.h[0] = (bf16_t)x; c.h[1] = (bf16_t)y;
  return c.u;
}

DEV uint2 pack4bf(float4 v) {
  uint2 r; r.x = pack2bf(v.x, v.y); r.y = pack2bf(v.z, v.w); return r;
}

// magic division: q = x/d via (x * M) >> 40, M = floor(2^40/d)+1.
// Valid for x < 2^40/d (largest x here ~1.3e5, smallest conv divisor 3).
DEV int mdiv(int x, unsigned long long magic) {
  return (int)(((unsigned long long)(unsigned)x * magic) >> 40);
}

struct GemmDesc {
  const float* A; const float* B; const float* bias; float* C;
  const uint16_t* Bpre;                  // bf16 pre-transposed weights [n][Kpad]
  int Kpad;
  int M, N, K, ldc;
  int mode, lda, ldbK, ldbN, relu, avec;
  long long aBatch, bBatch, cBatch;
  int Cin, IH, IW, OH, OW, KH, KW, stride, pad, chw, pix;
  long long imgStride;
  unsigned long long magicPix, magicOW, magicCin, magicKW;
};

#define MB 128
#define NB 64
#define KBB 64

union Frag { bf16x16 v; uint4 q[2]; };

template <int MODE, int BPRE>
__global__ __launch_bounds__(256) void k_gemm(GemmDesc g) {
  __shared__ uint16_t Asm[2][MB * KBB];
  __shared__ uint16_t Bsm[2][NB * KBB];
  int t = threadIdx.x;
  int wave = t >> 5, lane = t & 31;
  int half = lane >> 4, l15 = lane & 15;
  int bm = blockIdx.x * MB, bn = blockIdx.y * NB;
  long long zb = blockIdx.z;
  const float* A = g.A + zb * g.aBatch;
  const float* B = g.B + zb * g.bBatch;
  float*       C = g.C + zb * g.cBatch;
  int wm = (wave & 3) * 32, wn = (wave >> 2) * 32;
  int nK = (g.K + KBB - 1) / KBB;
  bool fullM = (bm + MB <= g.M);
  bool kfull = (g.K & 63) == 0;          // uniform: per-thread k always valid
  bool nfull = (g.N & 63) == 0;

  auto loadTiles = [&](int kt, int s) {
    // ------------------------- A tile: 128x64 bf16 -------------------------
    if (g.avec) {                        // float4 gather path (K%4==0)
      int kq = (t & 15) * 4;
      int mrow = t >> 4;                 // 0..15
      int k = kt * KBB + kq;
      int kh = 0, kw = 0, ci = 0;
      if constexpr (MODE != 0) {
        int tap = mdiv(k, g.magicCin); ci = k - tap * g.Cin;
        kh = mdiv(tap, g.magicKW);     kw = tap - kh * g.KW;
      }
      if (MODE == 0 && kfull && fullM) { // interior tile: unconditional b128
#pragma unroll
        for (int j = 0; j < 8; ++j) {
          int m = bm + j * 16 + mrow;
          float4 v = *(const float4*)&A[(long long)m * g.lda + k];
          *(uint2*)&Asm[s][(j * 16 + mrow) * KBB + kq] = pack4bf(v);
        }
      } else {
        bool kv = k < g.K;
#pragma unroll
        for (int j = 0; j < 8; ++j) {
          int m = bm + j * 16 + mrow;
          float4 v = make_float4(0.f, 0.f, 0.f, 0.f);
          if (kv && (fullM || m < g.M)) {
            if constexpr (MODE == 0) {
              v = *(const float4*)&A[(long long)m * g.lda + k];
            } else {
              int img = mdiv(m, g.magicPix);
              int r = m - img * g.pix;
              int oh = mdiv(r, g.magicOW);
              int ow = r - oh * g.OW;
              int ih, iw; bool ok;
              if constexpr (MODE == 1) {
                ih = oh + kh - g.pad; iw = ow + kw - g.pad;
                ok = (unsigned)ih < (unsigned)g.IH && (unsigned)iw < (unsigned)g.IW;
              } else {
                int th = oh - kh, tw = ow - kw;
                ok = (th >= 0) && (tw >= 0);
                if (g.stride == 2) { ok = ok && !((th | tw) & 1); ih = th >> 1; iw = tw >> 1; }
                else               { ih = th; iw = tw; }
                ok = ok && ih < g.IH && iw < g.IW;
              }
              if (ok)
                v = *(const float4*)&A[(long long)img * g.imgStride +
                                       (long long)(ih * g.IW + iw) * g.Cin + ci];
            }
          }
          *(uint2*)&Asm[s][(j * 16 + mrow) * KBB + kq] = pack4bf(v);
        }
      }
    } else {                             // scalar fallback (c1: CHW, Cin=3)
      int kk = t & 63;
      int tRow = t >> 6;
      int k = kt * KBB + kk;
      bool kv = k < g.K;
      int kh = 0, kw = 0, ci = 0;
      if constexpr (MODE != 0) {
        int tap = mdiv(k, g.magicCin); ci = k - tap * g.Cin;
        kh = mdiv(tap, g.magicKW);     kw = tap - kh * g.KW;
      }
#pragma unroll
      for (int j = 0; j < 32; ++j) {
        int m = bm + j * 4 + tRow;
        float v = 0.f;
        if (kv && m < g.M) {
          if constexpr (MODE == 0) {
            v = A[(long long)m * g.lda + k];
          } else {
            int img = mdiv(m, g.magicPix);
            int r = m - img * g.pix;
            int oh = mdiv(r, g.magicOW);
            int ow = r - oh * g.OW;
            int ih, iw; bool ok;
            if constexpr (MODE == 1) {
              ih = oh + kh - g.pad; iw = ow + kw - g.pad;
              ok = (unsigned)ih < (unsigned)g.IH && (unsigned)iw < (unsigned)g.IW;
            } else {
              int th = oh - kh, tw = ow - kw;
              ok = (th >= 0) && (tw >= 0);
              if (g.stride == 2) { ok = ok && !((th | tw) & 1); ih = th >> 1; iw = tw >> 1; }
              else               { ih = th; iw = tw; }
              ok = ok && ih < g.IH && iw < g.IW;
            }
            if (ok) {
              long long base = (long long)img * g.imgStride;
              v = g.chw ? A[base + ((long long)ci * g.IH + ih) * g.IW + iw]
                        : A[base + (long long)(ih * g.IW + iw) * g.Cin + ci];
            }
          }
        }
        Asm[s][(j * 4 + tRow) * KBB + kk] = f2bf(v);
      }
    }
    // ------------------------- B tile: 64x64 bf16 [n][k] -------------------
    if constexpr (BPRE) {                // bf16 weights, unconditional b128
      int kq = (t & 7) * 8;
      int nrow = t >> 3;                 // 0..31
#pragma unroll
      for (int j = 0; j < 2; ++j) {
        int n = j * 32 + nrow;
        uint4 v = *(const uint4*)&g.Bpre[(long long)(bn + n) * g.Kpad + kt * KBB + kq];
        *(uint4*)&Bsm[s][n * KBB + kq] = v;
      }
    } else {
      if (g.ldbK == 1) {                 // K-contiguous dynamic B (q.kT)
        int kq = (t & 15) * 4;
        int nr = t >> 4;
        int k = kt * KBB + kq;
        if (kfull && nfull) {            // uniform fast path
#pragma unroll
          for (int j = 0; j < 4; ++j) {
            int n = j * 16 + nr;
            float4 v = *(const float4*)&B[(long long)k + (long long)(bn + n) * g.ldbN];
            *(uint2*)&Bsm[s][n * KBB + kq] = pack4bf(v);
          }
        } else {
          bool kv = k < g.K;
#pragma unroll
          for (int j = 0; j < 4; ++j) {
            int n = j * 16 + nr;
            float4 v = make_float4(0.f, 0.f, 0.f, 0.f);
            if (kv && bn + n < g.N)
              v = *(const float4*)&B[(long long)k + (long long)(bn + n) * g.ldbN];
            *(uint2*)&Bsm[s][n * KBB + kq] = pack4bf(v);
          }
        }
      } else {                           // strided dynamic B (attn.v)
        int kk = t & 63;
        int tRow = t >> 6;
        int k = kt * KBB + kk;
        if (kfull && nfull) {
#pragma unroll
          for (int j = 0; j < 16; ++j) {
            int n = j * 4 + tRow;
            float v = B[(long long)k * g.ldbK + (long long)(bn + n) * g.ldbN];
            Bsm[s][n * KBB + kk] = f2bf(v);
          }
        } else {
          bool kv = k < g.K;
#pragma unroll
          for (int j = 0; j < 16; ++j) {
            int n = j * 4 + tRow;
            float v = 0.f;
            if (kv && bn + n < g.N)
              v = B[(long long)k * g.ldbK + (long long)(bn + n) * g.ldbN];
            Bsm[s][n * KBB + kk] = f2bf(v);
          }
        }
      }
      int kp = (kt + 2) * KBB;           // L2 prefetch hint
      if (kp < g.K)
        __builtin_prefetch(&B[(long long)kp * g.ldbK + (long long)bn * g.ldbN]);
    }
  };

  f32x8 acc[2][2];
  f32x8 zf = {0.f, 0.f, 0.f, 0.f, 0.f, 0.f, 0.f, 0.f};
#pragma unroll
  for (int i = 0; i < 2; ++i)
#pragma unroll
    for (int j = 0; j < 2; ++j) acc[i][j] = zf;

  loadTiles(0, 0);
  __syncthreads();
  for (int kt = 0; kt < nK; ++kt) {
    int s = kt & 1;
    if (kt + 1 < nK) loadTiles(kt + 1, s ^ 1);   // overlap global->LDS with WMMA
#pragma unroll
    for (int ks = 0; ks < 2; ++ks) {             // 8 WMMAs per barrier
      Frag af[2], bfr[2];
#pragma unroll
      for (int i = 0; i < 2; ++i) {
        const uint4* p = (const uint4*)&Asm[s][(wm + i * 16 + l15) * KBB];
        af[i].q[0] = p[ks * 4 + half]; af[i].q[1] = p[ks * 4 + 2 + half];
      }
#pragma unroll
      for (int i = 0; i < 2; ++i) {
        const uint4* p = (const uint4*)&Bsm[s][(wn + i * 16 + l15) * KBB];
        bfr[i].q[0] = p[ks * 4 + half]; bfr[i].q[1] = p[ks * 4 + 2 + half];
      }
#pragma unroll
      for (int i = 0; i < 2; ++i)
#pragma unroll
        for (int j = 0; j < 2; ++j)
          acc[i][j] = __builtin_amdgcn_wmma_f32_16x16x32_bf16(
              false, af[i].v, false, bfr[j].v, (short)0, acc[i][j], false, false);
    }
    __syncthreads();
  }

  // epilogue: bias + optional ReLU, f32 store (C layout: vgpr e -> M=e(+8))
  bool fullTile = fullM && (bn + NB <= g.N);
#pragma unroll
  for (int i = 0; i < 2; ++i) {
#pragma unroll
    for (int j = 0; j < 2; ++j) {
      int col = bn + wn + j * 16 + l15;
      if (!fullTile && col >= g.N) continue;
      float bv = g.bias ? g.bias[col] : 0.f;
#pragma unroll
      for (int e = 0; e < 8; ++e) {
        int row = bm + wm + i * 16 + e + half * 8;
        if (!fullTile && row >= g.M) continue;
        float v = acc[i][j][e] + bv;
        if (g.relu) v = v > 0.f ? v : 0.f;
        C[(long long)row * g.ldc + col] = v;
      }
    }
  }
}

// ---------------------------------------------------------------------------
// Weight prep: f32 [K][N] -> bf16 [n][Kpad] (zero padded), done once/launch
// ---------------------------------------------------------------------------
__global__ void k_prepw(const float* src, uint16_t* dst, int K, int N, int Kpad, int total) {
  int i = blockIdx.x * 256 + threadIdx.x;
  if (i >= total) return;
  int n = i / Kpad, k = i - n * Kpad;
  dst[i] = (k < K && n < N) ? f2bf(src[(long long)k * N + n]) : (uint16_t)0;
}

// ---------------------------------------------------------------------------
// Elementwise / reduction helpers
// ---------------------------------------------------------------------------
__global__ void k_slot_init(const float* mu, const float* ls, const float* nz,
                            float* out, int n) {
  int i = blockIdx.x * 256 + threadIdx.x;
  if (i < n) { int c = i & 63; out[i] = mu[c] + __expf(ls[c]) * nz[i]; }
}

__global__ void k_copy(const float* a, float* b, int n) {
  int i = blockIdx.x * 256 + threadIdx.x; if (i < n) b[i] = a[i];
}

__global__ void k_add(float* d, const float* s, int n) {
  int i = blockIdx.x * 256 + threadIdx.x; if (i < n) d[i] += s[i];
}

__global__ void k_posenc(float* x, const float* W, const float* b, int H, int Wd,
                         float invH, float invW, long long total) {
  long long i = (long long)blockIdx.x * 256 + threadIdx.x;
  if (i >= total) return;
  int c = (int)(i & 63); long long m = i >> 6;
  int pix = H * Wd; int r = (int)(m % pix);
  int h = r / Wd, w = r - h * Wd;
  float u = h * invH, v = w * invW;
  x[i] += u * W[c] + v * W[64 + c] + (1.f - u) * W[128 + c] + (1.f - v) * W[192 + c] + b[c];
}

__global__ void k_ln(const float* x, float* y, const float* g, const float* b, int rows) {
  int row = blockIdx.x * 8 + (threadIdx.x >> 5);
  int lane = threadIdx.x & 31;
  if (row >= rows) return;
  const float* xr = x + (long long)row * 64;
  float a = xr[lane], c = xr[lane + 32];
  float s = a + c;
  for (int o = 16; o > 0; o >>= 1) s += __shfl_xor(s, o, 32);
  float mean = s * (1.f / 64.f);
  float da = a - mean, dc = c - mean;
  float v = da * da + dc * dc;
  for (int o = 16; o > 0; o >>= 1) v += __shfl_xor(v, o, 32);
  float inv = rsqrtf(v * (1.f / 64.f) + 1e-5f);
  float* yr = y + (long long)row * 64;
  yr[lane]      = da * inv * g[lane] + b[lane];
  yr[lane + 32] = dc * inv * g[lane + 32] + b[lane + 32];
}

__global__ void k_softmax8(float* d, int N, float scale, float eps, long long total) {
  long long i = (long long)blockIdx.x * 256 + threadIdx.x;
  if (i >= total) return;
  int b = (int)(i / N); int n = (int)(i - (long long)b * N);
  float* p = d + (long long)b * 8 * N + n;
  float v[8]; float mx = -1e30f;
  for (int s = 0; s < 8; ++s) { v[s] = p[(long long)s * N] * scale; mx = fmaxf(mx, v[s]); }
  float sum = 0.f;
  for (int s = 0; s < 8; ++s) { v[s] = __expf(v[s] - mx); sum += v[s]; }
  float inv = 1.f / sum;
  for (int s = 0; s < 8; ++s) p[(long long)s * N] = v[s] * inv + eps;
}

__global__ void k_rowsum(const float* a, float* sums, int N) {
  __shared__ float red[256];
  const float* p = a + (long long)blockIdx.x * N;
  float s = 0.f;
  for (int i = threadIdx.x; i < N; i += 256) s += p[i];
  red[threadIdx.x] = s; __syncthreads();
  for (int o = 128; o > 0; o >>= 1) {
    if (threadIdx.x < o) red[threadIdx.x] += red[threadIdx.x + o];
    __syncthreads();
  }
  if (!threadIdx.x) sums[blockIdx.x] = red[0];
}

__global__ void k_divrow(float* a, const float* sums, int N, long long total) {
  long long i = (long long)blockIdx.x * 256 + threadIdx.x;
  if (i >= total) return;
  a[i] /= sums[i / N];
}

__global__ void k_gru(const float* gi, const float* gh, float* h, int rows) {
  int i = blockIdx.x * 256 + threadIdx.x;
  if (i >= rows * 64) return;
  int r = i >> 6, c = i & 63;
  const float* gir = gi + (long long)r * 192;
  const float* ghr = gh + (long long)r * 192;
  float rg = 1.f / (1.f + __expf(-(gir[c]       + ghr[c])));
  float zg = 1.f / (1.f + __expf(-(gir[64 + c]  + ghr[64 + c])));
  float ng = tanhf(gir[128 + c] + rg * ghr[128 + c]);
  h[i] = (1.f - zg) * ng + zg * h[i];
}

__global__ void k_mha(const float* qkv, float* o) {  // B=4,S=8,H=4,dh=16
  __shared__ float qs[8][16], ks[8][16], vs[8][16], sc[8][8], aw[8][8];
  int b = blockIdx.x >> 2, h = blockIdx.x & 3;
  int tid = threadIdx.x, s = tid >> 4, d = tid & 15;
  const float* row = qkv + (long long)(b * 8 + s) * 192 + h * 48 + d;
  qs[s][d] = row[0]; ks[s][d] = row[16]; vs[s][d] = row[32];
  __syncthreads();
  if (tid < 64) {
    int i = tid >> 3, j = tid & 7;
    float a = 0.f;
    for (int e = 0; e < 16; ++e) a += qs[i][e] * ks[j][e];
    sc[i][j] = a * 0.25f;
  }
  __syncthreads();
  if (tid < 8) {
    float mx = -1e30f;
    for (int j = 0; j < 8; ++j) mx = fmaxf(mx, sc[tid][j]);
    float e[8], sum = 0.f;
    for (int j = 0; j < 8; ++j) { e[j] = __expf(sc[tid][j] - mx); sum += e[j]; }
    for (int j = 0; j < 8; ++j) aw[tid][j] = e[j] / sum;
  }
  __syncthreads();
  float a = 0.f;
  for (int j = 0; j < 8; ++j) a += aw[s][j] * vs[j][d];
  o[(long long)(b * 8 + s) * 64 + h * 16 + d] = a;
}

__global__ void k_sb(const float* slots, float* sb_out, float* decIn,
                     const float* posW, const float* posB) {
  int i = blockIdx.x * 256 + threadIdx.x;
  if (i >= 32 * 64 * 64) return;
  int c = i & 63; int rest = i >> 6;
  int pix = rest & 63; int img = rest >> 6;
  int ph = pix >> 3, pw = pix & 7;
  float sv = slots[(long long)img * 64 + c];
  sb_out[i] = sv;
  float u = ph * (1.f / 7.f), v = pw * (1.f / 7.f);
  decIn[i] = sv + u * posW[c] + v * posW[64 + c] +
             (1.f - u) * posW[128 + c] + (1.f - v) * posW[192 + c] + posB[c];
}

__global__ void k_combine(const float* d6, float* out) {
  int i = blockIdx.x * 256 + threadIdx.x;
  if (i >= 4 * 128 * 128) return;
  int w = i & 127; int r = i >> 7; int h = r & 127; int b = r >> 7;
  const int P = 179 * 179;
  float alpha[8]; float mx = -1e30f;
  for (int k = 0; k < 8; ++k) {
    alpha[k] = d6[((long long)(b * 8 + k) * P + h * 179 + w) * 4 + 3];
    mx = fmaxf(mx, alpha[k]);
  }
  float sum = 0.f;
  for (int k = 0; k < 8; ++k) { alpha[k] = __expf(alpha[k] - mx); sum += alpha[k]; }
  float inv = 1.f / sum;
  const long long offRecons = 196608;
  const long long offMasks  = 196608 + 1572864;
  float rc[3] = {0.f, 0.f, 0.f};
  for (int k = 0; k < 8; ++k) {
    float mk = alpha[k] * inv;
    long long src = ((long long)(b * 8 + k) * P + h * 179 + w) * 4;
    long long rb  = offRecons + (((long long)(b * 8 + k) * 128 + h) * 128 + w) * 3;
    for (int c = 0; c < 3; ++c) {
      float rv = d6[src + c];
      out[rb + c] = rv;
      rc[c] += rv * mk;
    }
    out[offMasks + ((long long)(b * 8 + k) * 128 + h) * 128 + w] = mk;
  }
  for (int c = 0; c < 3; ++c)
    out[(((long long)b * 3 + c) * 128 + h) * 128 + w] = rc[c];
}

// ---------------------------------------------------------------------------
// Host orchestration
// ---------------------------------------------------------------------------
struct PW { const uint16_t* p; int kpad; };

static unsigned long long mk_magic(int d) {
  return d > 0 ? ((1ull << 40) / (unsigned)d + 1) : 1;
}

static void gemm_launch(hipStream_t st, GemmDesc g, int batches) {
  g.pix = g.OH * g.OW;
  g.magicPix = mk_magic(g.pix);
  g.magicOW  = mk_magic(g.OW);
  g.magicCin = mk_magic(g.Cin);
  g.magicKW  = mk_magic(g.KW);
  g.avec = (g.mode == 0) ? ((g.lda & 3) == 0 && (g.K & 3) == 0)
                         : (!g.chw && (g.Cin & 3) == 0);
  dim3 grid((g.M + MB - 1) / MB, (g.N + NB - 1) / NB, batches);
  if (g.Bpre) {
    if (g.mode == 0)      k_gemm<0, 1><<<grid, 256, 0, st>>>(g);
    else if (g.mode == 1) k_gemm<1, 1><<<grid, 256, 0, st>>>(g);
    else                  k_gemm<2, 1><<<grid, 256, 0, st>>>(g);
  } else {
    k_gemm<0, 0><<<grid, 256, 0, st>>>(g);
  }
}

static GemmDesc dense(const float* A, PW w, const float* bias, float* C,
                      int M, int N, int K, int lda, int ldc, int relu) {
  GemmDesc g{}; g.A = A; g.Bpre = w.p; g.Kpad = w.kpad; g.bias = bias; g.C = C;
  g.M = M; g.N = N; g.K = K; g.lda = lda; g.ldc = ldc; g.mode = 0; g.relu = relu;
  return g;
}

static GemmDesc denseDyn(const float* A, const float* B, float* C,
                         int M, int N, int K, int lda, int ldbK, int ldbN, int ldc) {
  GemmDesc g{}; g.A = A; g.B = B; g.C = C;
  g.M = M; g.N = N; g.K = K; g.lda = lda; g.ldbK = ldbK; g.ldbN = ldbN;
  g.ldc = ldc; g.mode = 0; return g;
}

static GemmDesc convg(const float* A, PW w, const float* bias, float* C,
                      int imgs, int Cin, int IH, int OH, int KH, int stride, int pad,
                      int chw, long long imgStride, int Cout, int ldc, int relu, int mode) {
  GemmDesc g{}; g.A = A; g.Bpre = w.p; g.Kpad = w.kpad; g.bias = bias; g.C = C;
  g.M = imgs * OH * OH; g.N = Cout; g.K = KH * KH * Cin;
  g.ldc = ldc; g.mode = mode; g.relu = relu;
  g.Cin = Cin; g.IH = IH; g.IW = IH; g.OH = OH; g.OW = OH;
  g.KH = KH; g.KW = KH; g.stride = stride; g.pad = pad; g.chw = chw;
  g.imgStride = imgStride; return g;
}

extern "C" void kernel_launch(void* const* d_in, const int* in_sizes, int n_in,
                              void* d_out, int out_size, void* d_ws, size_t ws_size,
                              hipStream_t stream) {
#define IN(i) ((const float*)d_in[i])
  // flattened param order (JAX pytree: dict keys sorted)
  const float* images = IN(0);
  const float *d1b=IN(1),*d1w=IN(2),*d2b=IN(3),*d2w=IN(4),*d3b=IN(5),*d3w=IN(6);
  const float *d4b=IN(7),*d4w=IN(8),*d5b=IN(9),*d5w=IN(10),*d6b=IN(11),*d6w=IN(12);
  const float *dposb=IN(13),*dposw=IN(14);
  const float *c1b=IN(15),*c1w=IN(16),*c2b=IN(17),*c2w=IN(18),*c3b=IN(19),*c3w=IN(20);
  const float *c4b=IN(21),*c4w=IN(22),*elnb=IN(23),*elng=IN(24);
  const float *em1b=IN(25),*em1w=IN(26),*em2b=IN(27),*em2w=IN(28);
  const float *eposb=IN(29),*eposw=IN(30);
  const float *f1b=IN(31),*f1w=IN(32),*f2b=IN(33),*f2w=IN(34);
  const float *n1b=IN(35),*n1g=IN(36),*n2b=IN(37),*n2g=IN(38);
  const float *ob=IN(39),*ow=IN(40),*qkvb=IN(41),*qkvw=IN(42);
  const float *ghb=IN(43),*ghw=IN(44),*gib=IN(45),*giw=IN(46);
  const float *kb=IN(47),*kw=IN(48),*logsig=IN(49);
  const float *sm1b=IN(50),*sm1w=IN(51),*sm2b=IN(52),*sm2w=IN(53),*mu=IN(54);
  const float *nfb=IN(55),*nfg=IN(56),*ninb=IN(57),*ning=IN(58);
  const float *nsb=IN(59),*nsg=IN(60),*qb=IN(61),*qw=IN(62),*vb=IN(63),*vw=IN(64);
  const float* noise = IN(65);
  float* out = (float*)d_out;

  // workspace bump allocator (floats)
  float* W = (float*)d_ws; size_t off = 0;
  auto alloc = [&](size_t n) { float* p = W + off; off += n; return p; };
  float* buf0 = alloc(4194304);           // B*128*128*64
  float* buf1 = alloc(4194304);
  float* buf2 = alloc(4194304);
  float* decP = alloc((size_t)177*177*64*4);
  float* decQ = alloc((size_t)173*173*64*4);
  float* d6o  = alloc((size_t)32*179*179*4);
  float* attn = alloc(4*8*16384);
  float* rows = alloc(64);
  float* slotInit = alloc(2048);
  float* slotsCur = alloc(2048);
  float* sTmp = alloc(2048);
  float* qBuf = alloc(2048);
  float* updB = alloc(2048);
  float* giB  = alloc(6144);
  float* ghB  = alloc(6144);
  float* mTmp = alloc(4096);
  float* sT2  = alloc(2048);
  float* qkvB = alloc(6144);
  float* mhaO = alloc(2048);
  float* oB   = alloc(2048);
  float* pT   = alloc(2048);
  float* fB   = alloc(8192);
  float* f2B  = alloc(2048);
  float* decIn = alloc(131072);
  uint16_t* wbase = (uint16_t*)alloc(600000);   // bf16 weight pool (~1 Mu16)
  (void)ws_size; (void)n_in; (void)in_sizes; (void)out_size;

  // -------- one-shot weight conversion: f32 [K][N] -> bf16 [n][Kpad] -------
  size_t woff = 0;
  auto prepw = [&](const float* src, int K, int N) -> PW {
    int Kpad = (K + 63) / 64 * 64;
    int Nal  = (N + 63) / 64 * 64;
    uint16_t* dst = wbase + woff; woff += (size_t)Kpad * Nal;
    int total = Kpad * Nal;
    k_prepw<<<(total + 255) / 256, 256, 0, stream>>>(src, dst, K, N, Kpad, total);
    return PW{dst, Kpad};
  };
  PW pc1 = prepw(c1w, 75, 64),   pc2 = prepw(c2w, 1600, 64);
  PW pc3 = prepw(c3w, 1600, 64), pc4 = prepw(c4w, 1600, 64);
  PW pm1 = prepw(em1w, 64, 64),  pm2 = prepw(em2w, 64, 64);
  PW pk  = prepw(kw, 64, 64),    pv  = prepw(vw, 64, 64),  pq = prepw(qw, 64, 64);
  PW pgi = prepw(giw, 64, 192),  pgh = prepw(ghw, 64, 192);
  PW ps1 = prepw(sm1w, 64, 128), ps2 = prepw(sm2w, 128, 64);
  PW pqkv= prepw(qkvw, 64, 192), po  = prepw(ow, 64, 64);
  PW pf1 = prepw(f1w, 64, 256),  pf2 = prepw(f2w, 256, 64);
  PW pd1 = prepw(d1w, 1600, 64), pd2 = prepw(d2w, 1600, 64);
  PW pd3 = prepw(d3w, 1600, 64), pd4 = prepw(d4w, 1600, 64);
  PW pd5 = prepw(d5w, 1600, 64), pd6 = prepw(d6w, 576, 4);

  const int M_ENC = 4 * 128 * 128;        // 65536
  const long long IMG_STRIDE = 4LL * 3 * 128 * 128;   // per-b stride (NCHW, T inner)

  k_slot_init<<<8, 256, 0, stream>>>(mu, logsig, noise, slotInit, 2048);

  for (int t = 0; t < 4; ++t) {
    // ---------------- encoder ----------------
    const float* imgT = images + (long long)t * 3 * 128 * 128;
    gemm_launch(stream, convg(imgT, pc1, c1b, buf0, 4, 3, 128, 128, 5, 1, 2, 1, IMG_STRIDE, 64, 64, 1, 1), 1);
    gemm_launch(stream, convg(buf0, pc2, c2b, buf1, 4, 64, 128, 128, 5, 1, 2, 0, 1048576, 64, 64, 1, 1), 1);
    gemm_launch(stream, convg(buf1, pc3, c3b, buf0, 4, 64, 128, 128, 5, 1, 2, 0, 1048576, 64, 64, 1, 1), 1);
    gemm_launch(stream, convg(buf0, pc4, c4b, buf1, 4, 64, 128, 128, 5, 1, 2, 0, 1048576, 64, 64, 1, 1), 1);
    k_posenc<<<(M_ENC*64+255)/256, 256, 0, stream>>>(buf1, eposw, eposb, 128, 128,
                                                     1.f/127.f, 1.f/127.f, (long long)M_ENC*64);
    k_ln<<<(M_ENC+7)/8, 256, 0, stream>>>(buf1, buf0, elng, elnb, M_ENC);
    gemm_launch(stream, dense(buf0, pm1, em1b, buf2, M_ENC, 64, 64, 64, 64, 1), 1);
    gemm_launch(stream, dense(buf2, pm2, em2b, buf1, M_ENC, 64, 64, 64, 64, 0), 1);

    // ---------------- slot attention ----------------
    k_ln<<<(M_ENC+7)/8, 256, 0, stream>>>(buf1, buf0, ning, ninb, M_ENC);      // x
    gemm_launch(stream, dense(buf0, pk, kb, buf2, M_ENC, 64, 64, 64, 64, 0), 1); // k
    gemm_launch(stream, dense(buf0, pv, vb, buf1, M_ENC, 64, 64, 64, 64, 0), 1); // v
    k_copy<<<8, 256, 0, stream>>>(slotInit, slotsCur, 2048);

    for (int it = 0; it < 3; ++it) {
      k_ln<<<4, 256, 0, stream>>>(slotsCur, sTmp, nsg, nsb, 32);
      gemm_launch(stream, dense(sTmp, pq, qb, qBuf, 32, 64, 64, 64, 64, 0), 1);
      { GemmDesc g = denseDyn(qBuf, buf2, attn, 8, 16384, 64, 64, 1, 64, 16384);
        g.aBatch = 512; g.bBatch = 1048576; g.cBatch = 131072;
        gemm_launch(stream, g, 4); }                                     // dots = q.kT
      k_softmax8<<<(4*16384+255)/256, 256, 0, stream>>>(attn, 16384, 0.125f, 1e-8f, 4LL*16384);
      k_rowsum<<<32, 256, 0, stream>>>(attn, rows, 16384);
      k_divrow<<<(32*16384+255)/256, 256, 0, stream>>>(attn, rows, 16384, 32LL*16384);
      { GemmDesc g = denseDyn(attn, buf1, updB, 8, 64, 16384, 16384, 64, 1, 64);
        g.aBatch = 131072; g.bBatch = 1048576; g.cBatch = 512;
        gemm_launch(stream, g, 4); }                                     // updates
      gemm_launch(stream, dense(updB, pgi, gib, giB, 32, 192, 64, 64, 192, 0), 1);
      gemm_launch(stream, dense(slotsCur, pgh, ghb, ghB, 32, 192, 64, 64, 192, 0), 1);
      k_gru<<<8, 256, 0, stream>>>(giB, ghB, slotsCur, 32);
      k_ln<<<4, 256, 0, stream>>>(slotsCur, sTmp, nfg, nfb, 32);
      gemm_launch(stream, dense(sTmp, ps1, sm1b, mTmp, 32, 128, 64, 64, 128, 1), 1);
      gemm_launch(stream, dense(mTmp, ps2, sm2b, sT2, 32, 64, 128, 128, 64, 0), 1);
      k_add<<<8, 256, 0, stream>>>(slotsCur, sT2, 2048);
    }

    // ---------------- predictor (not needed after last step) -------------
    if (t < 3) {
      gemm_launch(stream, dense(slotsCur, pqkv, qkvb, qkvB, 32, 192, 64, 64, 192, 0), 1);
      k_mha<<<16, 128, 0, stream>>>(qkvB, mhaO);
      gemm_launch(stream, dense(mhaO, po, ob, oB, 32, 64, 64, 64, 64, 0), 1);
      k_add<<<8, 256, 0, stream>>>(oB, slotsCur, 2048);
      k_ln<<<4, 256, 0, stream>>>(oB, pT, n1g, n1b, 32);
      gemm_launch(stream, dense(pT, pf1, f1b, fB, 32, 256, 64, 64, 256, 1), 1);
      gemm_launch(stream, dense(fB, pf2, f2b, f2B, 32, 64, 256, 256, 64, 0), 1);
      k_add<<<8, 256, 0, stream>>>(f2B, pT, 2048);
      k_ln<<<4, 256, 0, stream>>>(f2B, slotInit, n2g, n2b, 32);
    }
  }

  // ---------------- decoder (final timestep only) ----------------
  k_sb<<<512, 256, 0, stream>>>(slotsCur, out + 2293760, decIn, dposw, dposb);
  for (int gidx = 0; gidx < 8; ++gidx) {   // groups of 4 slot-images
    const float* in0 = decIn + (long long)gidx * 4 * 4096;
    gemm_launch(stream, convg(in0, pd1, d1b, decP, 4, 64,   8,  19, 5, 2, 0, 0,    4096, 64, 64, 1, 2), 1);
    gemm_launch(stream, convg(decP, pd2, d2b, decQ, 4, 64,  19,  41, 5, 2, 0, 0,   23104, 64, 64, 1, 2), 1);
    gemm_launch(stream, convg(decQ, pd3, d3b, decP, 4, 64,  41,  85, 5, 2, 0, 0,  107584, 64, 64, 1, 2), 1);
    gemm_launch(stream, convg(decP, pd4, d4b, decQ, 4, 64,  85, 173, 5, 2, 0, 0,  462400, 64, 64, 1, 2), 1);
    gemm_launch(stream, convg(decQ, pd5, d5b, decP, 4, 64, 173, 177, 5, 1, 0, 0, 1915456, 64, 64, 1, 2), 1);
    gemm_launch(stream, convg(decP, pd6, d6b, d6o + (long long)gidx*4*179*179*4,
                              4, 64, 177, 179, 3, 1, 0, 0, 2005056, 4, 4, 0, 2), 1);
  }
  k_combine<<<(4*128*128+255)/256, 256, 0, stream>>>(d6o, out);
#undef IN
}
